// GCNModel_31817117729398
// MI455X (gfx1250) — matrix operand order repaired and checked by reading, not compiled, not run on hardware
//
#include <hip/hip_runtime.h>
#include <math.h>

// ---------- types for WMMA ----------
typedef __bf16 v16bf __attribute__((ext_vector_type(16)));
typedef float  v8f   __attribute__((ext_vector_type(8)));

// Hardware f32 global atomic add, non-returning (GLOBAL_ATOMIC_ADD_F32).
// Forced via inline asm so it cannot lower to a CAS loop: the edge-scatter
// kernel is the bandwidth-dominant phase (~1.8 GB of atomic traffic) and
// must use the single-instruction L2 atomic path.
__device__ __forceinline__ void atomAddF(float* p, float v) {
#if defined(__gfx1250__)
    asm volatile("global_atomic_add_f32 %0, %1, off" ::"v"(p), "v"(v) : "memory");
#else
    atomicAdd(p, v);
#endif
}

// -------------------------------------------------------------------------
// Pack a fp32 weight matrix W [128 x ncols] (row-major, K x N) into bf16
// B-fragment order for v_wmma_f32_16x16x32_bf16:
//   frag element j of lane L in tile (nt, kt):
//     K = kt*32 + (L<16 ? 0 : 16) + j,  N = nt*16 + (L & 15)
// Layout: out[((nt*4 + kt)*32 + lane)*16 + j], zero-padded for N >= ncols.
// -------------------------------------------------------------------------
__global__ void pack_w_kernel(const float* __restrict__ W, int ncols, int ntiles,
                              __bf16* __restrict__ out) {
    int idx = blockIdx.x * blockDim.x + threadIdx.x;
    int total = ntiles * 4 * 32 * 16;
    if (idx >= total) return;
    int j    = idx & 15;
    int lane = (idx >> 4) & 31;
    int kt   = (idx >> 9) & 3;
    int nt   = idx >> 11;
    int K    = kt * 32 + ((lane & 16) ? 16 : 0) + j;
    int col  = nt * 16 + (lane & 15);
    float v  = (col < ncols) ? W[K * ncols + col] : 0.0f;
    out[idx] = (__bf16)v;
}

// -------------------------------------------------------------------------
// H[M x (NT*16)] = A[M x 128] * Wfrag   (bf16 WMMA, f32 accumulate)
// One wave (wave32) per 16x16 output tile; K = 128 = 4 steps of 32.
// -------------------------------------------------------------------------
template <int NT>
__global__ void gemm_wmma_kernel(const float* __restrict__ A,
                                 const __bf16* __restrict__ Bf,
                                 float* __restrict__ H, int M) {
    int wid  = (blockIdx.x * blockDim.x + threadIdx.x) >> 5;
    int lane = threadIdx.x & 31;
    int tiles = (M >> 4) * NT;
    if (wid >= tiles) return;
    int mt = wid / NT;
    int nt = wid % NT;

    // A-fragment addressing: lanes 0-15 -> K offs {0..15}; 16-31 -> {16..31}
    int row   = mt * 16 + (lane & 15);
    int khalf = (lane & 16) ? 16 : 0;
    const float* ap = A + (size_t)row * 128 + khalf;
    const v16bf* bp = reinterpret_cast<const v16bf*>(Bf) + (size_t)(nt * 4) * 32 + lane;

    v8f acc = {};
#pragma unroll
    for (int kk = 0; kk < 4; ++kk) {
        const float* p = ap + kk * 32;
        float4 f0 = *(const float4*)(p);
        float4 f1 = *(const float4*)(p + 4);
        float4 f2 = *(const float4*)(p + 8);
        float4 f3 = *(const float4*)(p + 12);
        v16bf a;
        a[0]  = (__bf16)f0.x; a[1]  = (__bf16)f0.y; a[2]  = (__bf16)f0.z; a[3]  = (__bf16)f0.w;
        a[4]  = (__bf16)f1.x; a[5]  = (__bf16)f1.y; a[6]  = (__bf16)f1.z; a[7]  = (__bf16)f1.w;
        a[8]  = (__bf16)f2.x; a[9]  = (__bf16)f2.y; a[10] = (__bf16)f2.z; a[11] = (__bf16)f2.w;
        a[12] = (__bf16)f3.x; a[13] = (__bf16)f3.y; a[14] = (__bf16)f3.z; a[15] = (__bf16)f3.w;
        v16bf b = bp[kk * 32];
        acc = __builtin_amdgcn_wmma_f32_16x16x32_bf16(
            /*neg_a=*/false, a, /*neg_b=*/false, b,
            /*c_mod=*/(short)0, acc, /*reuse_a=*/false, /*reuse_b=*/false);
    }

    // C/D layout: VGPR r -> M = mt*16 + r (+8 for lanes 16-31), N = nt*16 + (lane&15)
    int col   = nt * 16 + (lane & 15);
    int rbase = mt * 16 + ((lane & 16) ? 8 : 0);
    float* hp = H + (size_t)rbase * (NT * 16) + col;
#pragma unroll
    for (int r = 0; r < 8; ++r) hp[(size_t)r * (NT * 16)] = acc[r];
}

// ---------- degree / norm ----------
__global__ void deg_init_kernel(float* deg, int n) {
    int i = blockIdx.x * blockDim.x + threadIdx.x;
    if (i < n) deg[i] = 1.0f;  // self-loop
}
__global__ void deg_edges_kernel(const int* __restrict__ dst, int E, float* deg) {
    int e = blockIdx.x * blockDim.x + threadIdx.x;
    if (e < E) atomAddF(&deg[dst[e]], 1.0f);
}
__global__ void dinv_kernel(const float* __restrict__ deg, float* __restrict__ dinv, int n) {
    int i = blockIdx.x * blockDim.x + threadIdx.x;
    if (i < n) dinv[i] = rsqrtf(deg[i]);  // deg >= 1 always
}

// ---------- aggregation: G[i] = dinv[i]^2 * H[i]  (self-loop init) ----------
template <int CH>
__global__ void self_init_kernel(const float* __restrict__ H, const float* __restrict__ dinv,
                                 float* __restrict__ G, int n) {
    int idx = blockIdx.x * blockDim.x + threadIdx.x;
    int total = n * (CH / 4);
    if (idx >= total) return;
    int node = idx / (CH / 4);
    int c4   = (idx % (CH / 4)) * 4;
    float w  = dinv[node] * dinv[node];
    float4 v = *(const float4*)(H + (size_t)node * CH + c4);
    float4 o = make_float4(v.x * w, v.y * w, v.z * w, v.w * w);
    *(float4*)(G + (size_t)node * CH + c4) = o;
}

// ---------- aggregation: G[dst] += dinv[s]*dinv[d] * H[src]  ----------
template <int CH>
__global__ void edge_scatter_kernel(const int* __restrict__ src, const int* __restrict__ dst,
                                    const float* __restrict__ dinv, const float* __restrict__ H,
                                    float* __restrict__ G, int E) {
    constexpr int PER = CH / 4;       // float4 slots per edge
    constexpr int EPW = 32 / PER;     // edges per wave
    int wid  = (blockIdx.x * blockDim.x + threadIdx.x) >> 5;
    int lane = threadIdx.x & 31;
    int e = wid * EPW + lane / PER;
    if (e >= E) return;
    int s = src[e], d = dst[e];
    float w = dinv[s] * dinv[d];
    int c4 = (lane % PER) * 4;
    float4 v = *(const float4*)(H + (size_t)s * CH + c4);
    float* g = G + (size_t)d * CH + c4;
    atomAddF(g + 0, v.x * w);
    atomAddF(g + 1, v.y * w);
    atomAddF(g + 2, v.z * w);
    atomAddF(g + 3, v.w * w);
}

// ---------- bias + relu ----------
__global__ void bias_relu_kernel(const float* __restrict__ G, const float* __restrict__ b,
                                 float* __restrict__ X, int n, int CH) {
    int idx = blockIdx.x * blockDim.x + threadIdx.x;
    if (idx >= n * CH) return;
    float v = G[idx] + b[idx % CH];
    X[idx] = fmaxf(v, 0.0f);
}

// ---------- mean-pool + log_softmax ----------
__global__ void pool_init_kernel(float* pooled) {
    if (threadIdx.x < 16 && blockIdx.x == 0) pooled[threadIdx.x] = 0.0f;
}
__global__ void pool_partial_kernel(const float* __restrict__ G, int n, int ncls,
                                    float* __restrict__ pooled) {
    float acc[16];
#pragma unroll
    for (int c = 0; c < 16; ++c) acc[c] = 0.0f;
    for (int i = blockIdx.x * blockDim.x + threadIdx.x; i < n; i += gridDim.x * blockDim.x)
        for (int c = 0; c < ncls; ++c) acc[c] += G[(size_t)i * 16 + c];
    for (int c = 0; c < ncls; ++c) atomAddF(&pooled[c], acc[c]);
}
__global__ void finalize_kernel(const float* __restrict__ pooled, const float* __restrict__ b3,
                                float* __restrict__ out, float invN, int ncls) {
    if (threadIdx.x != 0 || blockIdx.x != 0) return;
    float v[16], mx = -1e30f;
    for (int c = 0; c < ncls; ++c) { v[c] = pooled[c] * invN + b3[c]; mx = fmaxf(mx, v[c]); }
    float s = 0.0f;
    for (int c = 0; c < ncls; ++c) s += __expf(v[c] - mx);
    float l = __logf(s) + mx;
    for (int c = 0; c < ncls; ++c) out[c] = v[c] - l;
}

static inline int cdiv(long long a, int b) { return (int)((a + b - 1) / b); }

extern "C" void kernel_launch(void* const* d_in, const int* in_sizes, int n_in,
                              void* d_out, int out_size, void* d_ws, size_t ws_size,
                              hipStream_t stream) {
    const float* x  = (const float*)d_in[0];
    const int*   ei = (const int*)d_in[1];
    const float* W1 = (const float*)d_in[2];
    const float* b1 = (const float*)d_in[3];
    const float* W2 = (const float*)d_in[4];
    const float* b2 = (const float*)d_in[5];
    const float* W3 = (const float*)d_in[6];
    const float* b3 = (const float*)d_in[7];
    float* out = (float*)d_out;

    const int CH   = 128;
    const int n    = in_sizes[0] / CH;   // 50000 (divisible by 16)
    const int E    = in_sizes[1] / 2;    // 800000
    const int ncls = in_sizes[7];        // 10
    const int* src = ei;
    const int* dst = ei + E;

    // ---- workspace layout (256B aligned regions) ----
    char* ws = (char*)d_ws;
    size_t off = 0;
    auto alloc = [&](size_t bytes) -> void* {
        void* p = ws + off;
        off += (bytes + 255) & ~(size_t)255;
        return p;
    };
    __bf16* wf1   = (__bf16*)alloc((size_t)8 * 4 * 32 * 16 * 2);
    __bf16* wf2   = (__bf16*)alloc((size_t)8 * 4 * 32 * 16 * 2);
    __bf16* wf3   = (__bf16*)alloc((size_t)1 * 4 * 32 * 16 * 2);
    float*  deg   = (float*)alloc((size_t)n * 4);
    float*  dinv  = (float*)alloc((size_t)n * 4);
    float*  Abuf  = (float*)alloc((size_t)n * CH * 4);
    float*  Hbuf  = (float*)alloc((size_t)n * CH * 4);
    float*  Gbuf  = (float*)alloc((size_t)n * CH * 4);
    float*  pooled = (float*)alloc(64);

    const int B = 256;

    // ---- pre-swizzle weights into bf16 B-fragment order ----
    pack_w_kernel<<<cdiv(8 * 2048, B), B, 0, stream>>>(W1, CH,   8, wf1);
    pack_w_kernel<<<cdiv(8 * 2048, B), B, 0, stream>>>(W2, CH,   8, wf2);
    pack_w_kernel<<<cdiv(1 * 2048, B), B, 0, stream>>>(W3, ncls, 1, wf3);

    // ---- degrees / symmetric norm ----
    deg_init_kernel<<<cdiv(n, B), B, 0, stream>>>(deg, n);
    deg_edges_kernel<<<cdiv(E, B), B, 0, stream>>>(dst, E, deg);
    dinv_kernel<<<cdiv(n, B), B, 0, stream>>>(deg, dinv, n);

    // ---- layer 1 ----
    {
        int tiles = (n / 16) * 8;
        gemm_wmma_kernel<8><<<cdiv((long long)tiles * 32, B), B, 0, stream>>>(x, wf1, Hbuf, n);
        self_init_kernel<128><<<cdiv((long long)n * 32, B), B, 0, stream>>>(Hbuf, dinv, Gbuf, n);
        edge_scatter_kernel<128><<<cdiv((long long)E * 32, B), B, 0, stream>>>(src, dst, dinv, Hbuf, Gbuf, E);
        bias_relu_kernel<<<cdiv((long long)n * CH, B), B, 0, stream>>>(Gbuf, b1, Abuf, n, CH);
    }
    // ---- layer 2 ----
    {
        int tiles = (n / 16) * 8;
        gemm_wmma_kernel<8><<<cdiv((long long)tiles * 32, B), B, 0, stream>>>(Abuf, wf2, Hbuf, n);
        self_init_kernel<128><<<cdiv((long long)n * 32, B), B, 0, stream>>>(Hbuf, dinv, Gbuf, n);
        edge_scatter_kernel<128><<<cdiv((long long)E * 32, B), B, 0, stream>>>(src, dst, dinv, Hbuf, Gbuf, E);
        bias_relu_kernel<<<cdiv((long long)n * CH, B), B, 0, stream>>>(Gbuf, b2, Abuf, n, CH);
    }
    // ---- layer 3 (N padded 10 -> 16; 8 edges per wave) ----
    {
        int tiles = (n / 16) * 1;
        gemm_wmma_kernel<1><<<cdiv((long long)tiles * 32, B), B, 0, stream>>>(Abuf, wf3, Hbuf, n);
        self_init_kernel<16><<<cdiv((long long)n * 4, B), B, 0, stream>>>(Hbuf, dinv, Gbuf, n);
        int waves = cdiv(E, 8);
        edge_scatter_kernel<16><<<cdiv((long long)waves * 32, B), B, 0, stream>>>(src, dst, dinv, Hbuf, Gbuf, E);
    }

    // ---- mean pool + bias + log_softmax ----
    pool_init_kernel<<<1, 32, 0, stream>>>(pooled);
    pool_partial_kernel<<<256, B, 0, stream>>>(Gbuf, n, ncls, pooled);
    finalize_kernel<<<1, 32, 0, stream>>>(pooled, b3, out, 1.0f / (float)n, ncls);
}
